// SimplifiedLinearAttention_72834055405758
// MI455X (gfx1250) — matrix operand order repaired and checked by reading, not compiled, not run on hardware
//
#include <hip/hip_runtime.h>
#include <hip/hip_bf16.h>

// SimplifiedLinearAttention for MI455X (gfx1250), wave32 + WMMA + TDM.
// Shapes: N=2, H=8, S=2048, D=64.  Outputs: context [16,2048,64] fp32
// followed by attention_weights [16,2048,2048] fp32 (268 MB -> bandwidth bound).
//
// Max-free softmax: post-presoftmax scores are provably in [0,64], so exp()
// cannot overflow fp32 and the row-max subtraction is mathematically a no-op.
// K/V tiles are double-buffered in LDS (TDM DMA for K when available), with a
// single workgroup barrier per 64-row k-tile.

#define SEQ 2048
#define DIM 64
#define NH  16          // N*H
#define QT  128         // q rows per workgroup (8 waves x 16 rows)
#define KT  64          // kv rows per pipeline stage
#define SW  (SEQ/32)    // mask words per row
#define NT  (SEQ/KT)    // 32 pipeline iterations per pass

typedef __attribute__((ext_vector_type(16))) _Float16 v16h;
typedef __attribute__((ext_vector_type(8)))  _Float16 v8h;
typedef __attribute__((ext_vector_type(8)))  float    v8f;

typedef unsigned int u32x4 __attribute__((ext_vector_type(4)));
typedef int          i32x4 __attribute__((ext_vector_type(4)));
typedef int          i32x8 __attribute__((ext_vector_type(8)));

#if defined(__has_builtin)
#if __has_builtin(__builtin_amdgcn_tensor_load_to_lds) && __has_builtin(__builtin_amdgcn_s_wait_tensorcnt)
#define HAVE_TDM 1
#endif
#endif

// ---------------------------------------------------------------------------
// Pre-pass 1: Q' = softmax(Q, axis=2)  (over S, per (n,h,d) column) -> fp16
__launch_bounds__(256)
__global__ void softmax_q_kernel(const float* __restrict__ Q,
                                 _Float16* __restrict__ Qh) {
  const int bh = blockIdx.x;
  const int d  = threadIdx.x & 63;
  const int sg = threadIdx.x >> 6;          // 0..3
  const float* base = Q + (size_t)bh * SEQ * DIM;
  __shared__ float red[4][64];

  float mx = -__builtin_inff();
  for (int i = 0; i < SEQ / 4; ++i)
    mx = fmaxf(mx, base[(size_t)(sg * (SEQ/4) + i) * DIM + d]);
  red[sg][d] = mx;
  __syncthreads();
  mx = fmaxf(fmaxf(red[0][d], red[1][d]), fmaxf(red[2][d], red[3][d]));
  __syncthreads();

  float sum = 0.f;
  for (int i = 0; i < SEQ / 4; ++i)
    sum += __expf(base[(size_t)(sg * (SEQ/4) + i) * DIM + d] - mx);
  red[sg][d] = sum;
  __syncthreads();
  sum = red[0][d] + red[1][d] + red[2][d] + red[3][d];
  const float inv = 1.f / sum;

  _Float16* ob = Qh + (size_t)bh * SEQ * DIM;
  for (int i = 0; i < SEQ / 4; ++i) {
    size_t off = (size_t)(sg * (SEQ/4) + i) * DIM + d;
    ob[off] = (_Float16)(__expf(base[off] - mx) * inv);
  }
}

// ---------------------------------------------------------------------------
// Pre-pass 2: K' = softmax(K, axis=1) (over H=8, stride S*D) -> fp16
__launch_bounds__(256)
__global__ void softmax_k_kernel(const float* __restrict__ K,
                                 _Float16* __restrict__ Kh) {
  const int t = blockIdx.x * 256 + threadIdx.x;     // over N*S*D = 262144
  const int n = t >> 17;                            // / (S*D = 131072)
  const int r = t & (SEQ * DIM - 1);
  const float* base = K + ((size_t)n * 8) * SEQ * DIM + r;
  float v[8];
  float mx = -__builtin_inff();
#pragma unroll
  for (int h = 0; h < 8; ++h) { v[h] = base[(size_t)h * SEQ * DIM]; mx = fmaxf(mx, v[h]); }
  float sum = 0.f;
#pragma unroll
  for (int h = 0; h < 8; ++h) { v[h] = __expf(v[h] - mx); sum += v[h]; }
  const float inv = 1.f / sum;
  _Float16* ob = Kh + ((size_t)n * 8) * SEQ * DIM + r;
#pragma unroll
  for (int h = 0; h < 8; ++h) ob[(size_t)h * SEQ * DIM] = (_Float16)(v[h] * inv);
}

// ---------------------------------------------------------------------------
// Pre-pass 3: V -> fp16
__launch_bounds__(256)
__global__ void convert_v_kernel(const float* __restrict__ V,
                                 _Float16* __restrict__ Vh) {
  const int t = blockIdx.x * 256 + threadIdx.x;     // NH*S*D = 2,097,152
  Vh[t] = (_Float16)V[t];
}

// ---------------------------------------------------------------------------
// Pre-pass 4: pack attn_mask (S,S) int32 -> 1 bit/entry (512 KB, L2 resident).
__launch_bounds__(256)
__global__ void pack_mask_kernel(const int* __restrict__ mask,
                                 unsigned* __restrict__ bits) {
  const int t = blockIdx.x * 256 + threadIdx.x;     // over S*S bits
  unsigned long long b = __ballot(mask[t] != 0);    // wave32: low 32 bits
  if ((threadIdx.x & 31) == 0) bits[t >> 5] = (unsigned)b;
}

// ---------------------------------------------------------------------------
#ifdef HAVE_TDM
// Issue a TDM 2D tile load: rows x 64 fp16, row stride 64, into LDS.
// D# built per CDNA5 ISA ch.8 (group0: count/lds/global/type, group1: dims).
__device__ __forceinline__ void tdm_load_tile(unsigned lds_byte_addr,
                                              const _Float16* gptr, int rows) {
  unsigned long long ga = (unsigned long long)(size_t)gptr;
  u32x4 g0;
  g0[0] = 1u;                                     // count=1, user descriptor
  g0[1] = lds_byte_addr;                          // lds_addr
  g0[2] = (unsigned)(ga & 0xFFFFFFFFu);           // global_addr[31:0]
  g0[3] = (unsigned)((ga >> 32) & 0x01FFFFFFu)    // global_addr[56:32]
        | (2u << 30);                             // type = 2 ("image")
  i32x8 g1;
  g1[0] = (int)(1u << 16);                        // data_size=1 (2B), wg_mask=0
  g1[1] = (int)((unsigned)DIM << 16);             // tensor_dim0[15:0]=64
  g1[2] = (int)((unsigned)SEQ << 16);             // tensor_dim1[15:0]=2048
  g1[3] = (int)((unsigned)DIM << 16);             // tile_dim0=64
  g1[4] = (int)(unsigned)rows;                    // tile_dim1=rows, tile_dim2=0
  g1[5] = (int)DIM;                               // tensor_dim0_stride=64
  g1[6] = 0;
  g1[7] = 0;
  i32x4 z4 = {0, 0, 0, 0};
#if __clang_major__ >= 23
  i32x8 z8 = {0, 0, 0, 0, 0, 0, 0, 0};
  __builtin_amdgcn_tensor_load_to_lds(g0, g1, z4, z4, z8, 0);
#else
  __builtin_amdgcn_tensor_load_to_lds(g0, g1, z4, z4, 0);
#endif
}
#endif

// ---------------------------------------------------------------------------
// Main attention kernel. Grid: (S/QT, NH). 256 threads = 8 wave32.
// Wave w owns q rows q0 + w*16 .. +15.  Double-buffered 64-row kv tiles,
// one barrier per tile; TDM DMA for K overlaps with compute.
__launch_bounds__(256)
__global__ void attn_kernel(const _Float16* __restrict__ Qh,
                            const _Float16* __restrict__ Kh,
                            const _Float16* __restrict__ Vh,
                            const unsigned* __restrict__ mbits,
                            float* __restrict__ ctx,    // [NH,S,D]
                            float* __restrict__ wout) { // [NH,S,S]
  __shared__ __align__(32) _Float16 sQ[QT * DIM];        // 16 KB
  __shared__ __align__(32) _Float16 sK[2][KT * DIM];     // 16 KB (K tiles, dbl-buf)
  __shared__ __align__(32) _Float16 sVt[2][DIM * KT];    // 16 KB (V^T [d][kv], dbl-buf)
  __shared__ __align__(32) _Float16 sP[8][16 * 32];      //  8 KB (per-wave P staging)

  const int tid  = threadIdx.x;
  const int w    = tid >> 5;
  const int lane = tid & 31;
  const int hb   = lane >> 4;     // half-wave selector (layout K/M split)
  const int ln   = lane & 15;
  const int bh   = blockIdx.y;
  const int q0   = blockIdx.x * QT;

  // ---- stage Q tile (128 x 64 fp16) ----
  {
    const uint4* src = (const uint4*)(Qh + ((size_t)bh * SEQ + q0) * DIM);
    uint4* dst = (uint4*)sQ;
    for (int i = tid; i < QT * DIM / 8; i += 256) dst[i] = src[i];
  }
  __syncthreads();

  // ---- per-wave A operands (Q rows; 16x32 A operand, k order shared with B) ----
  // lane (hb,ln): row M = ln; element j: K = hb*16 + j (contiguous).
  v16h aQ[2];
  {
    const _Float16* qrow = sQ + (w * 16 + ln) * DIM;
#pragma unroll
    for (int b = 0; b < 2; ++b) {
      v8h lo  = *(const v8h*)(qrow + b * 32 + hb * 16);
      v8h hi8 = *(const v8h*)(qrow + b * 32 + hb * 16 + 8);
      v16h a;
#pragma unroll
      for (int j = 0; j < 8; ++j) { a[j] = lo[j]; a[8 + j] = hi8[j]; }
      aQ[b] = a;
    }
  }

  // ===================== PASS 1: row sums of exp(scores) ==================
  float rsum[8];
#pragma unroll
  for (int v = 0; v < 8; ++v) rsum[v] = 0.f;

  // prologue: stage K tile 0 into buffer 0
#ifdef HAVE_TDM
  if (w == 0) tdm_load_tile((unsigned)(size_t)sK[0], Kh + (size_t)bh * SEQ * DIM, KT);
#else
  {
    const uint4* src = (const uint4*)(Kh + (size_t)bh * SEQ * DIM);
    uint4* dst = (uint4*)sK[0];
    for (int i = tid; i < KT * DIM / 8; i += 256) dst[i] = src[i];
  }
#endif

  for (int t = 0; t < NT; ++t) {
    const int cur = t & 1;
#ifdef HAVE_TDM
    if (w == 0) __builtin_amdgcn_s_wait_tensorcnt(0);  // tile t landed
#endif
    __syncthreads();
    if (t + 1 < NT) {  // kick off tile t+1 DMA while computing tile t
#ifdef HAVE_TDM
      if (w == 0)
        tdm_load_tile((unsigned)(size_t)sK[cur ^ 1],
                      Kh + ((size_t)bh * SEQ + (t + 1) * KT) * DIM, KT);
#else
      const uint4* src = (const uint4*)(Kh + ((size_t)bh * SEQ + (t + 1) * KT) * DIM);
      uint4* dst = (uint4*)sK[cur ^ 1];
      for (int i = tid; i < KT * DIM / 8; i += 256) dst[i] = src[i];
#endif
    }

    unsigned mw[8][2];  // 64 mask bits per row for this tile
#pragma unroll
    for (int v = 0; v < 8; ++v) {
      const int q = q0 + w * 16 + v + 8 * hb;
      uint2 mq = *(const uint2*)&mbits[(size_t)q * SW + t * 2];
      mw[v][0] = mq.x; mw[v][1] = mq.y;
    }

    const _Float16* kt = sK[cur];
#pragma unroll
    for (int half = 0; half < 2; ++half) {
#pragma unroll
      for (int sub = 0; sub < 2; ++sub) {
        v8f c = {};
#pragma unroll
        for (int b = 0; b < 2; ++b) {
          // B 32x16 layout: lane col N=ln, element j: K = hb*16 + j (contiguous)
          v16h bb = *(const v16h*)(kt + (half * 32 + sub * 16 + ln) * DIM + b * 32 + hb * 16);
          c = __builtin_amdgcn_wmma_f32_16x16x32_f16(false, aQ[b], false, bb,
                                                     (short)0, c, false, false);
        }
#pragma unroll
        for (int v = 0; v < 8; ++v) {
          const int bit = (mw[v][half] >> (sub * 16 + ln)) & 1;
          rsum[v] += bit ? __expf(c[v]) : 0.f;   // scores in [0,64]: no overflow
        }
      }
    }
  }

  // one cross-lane reduction per row, after the whole k loop
  float rinv[8];
#pragma unroll
  for (int v = 0; v < 8; ++v) {
    float s = rsum[v];
    s += __shfl_xor(s, 1);
    s += __shfl_xor(s, 2);
    s += __shfl_xor(s, 4);
    s += __shfl_xor(s, 8);
    rinv[v] = (s > 0.f) ? (1.f / s) : 0.f;
  }

  // ===================== PASS 2: weights out + P@V =====================
  v8f acc[4];
#pragma unroll
  for (int db = 0; db < 4; ++db) { v8f z = {}; acc[db] = z; }

  __syncthreads();  // pass-1 reads of sK complete before restaging

  // prologue: stage K+V tile 0 into buffer 0
#ifdef HAVE_TDM
  if (w == 0) tdm_load_tile((unsigned)(size_t)sK[0], Kh + (size_t)bh * SEQ * DIM, KT);
#else
  {
    const uint4* src = (const uint4*)(Kh + (size_t)bh * SEQ * DIM);
    uint4* dst = (uint4*)sK[0];
    for (int i = tid; i < KT * DIM / 8; i += 256) dst[i] = src[i];
  }
#endif
  {
    const int r  = tid >> 2;            // kv row 0..63
    const int dg = (tid & 3) * 16;      // d group
    const _Float16* vsrc = Vh + ((size_t)bh * SEQ + r) * DIM + dg;
#pragma unroll
    for (int i = 0; i < 16; ++i) sVt[0][(dg + i) * KT + r] = vsrc[i];
  }

  for (int t = 0; t < NT; ++t) {
    const int cur = t & 1;
#ifdef HAVE_TDM
    if (w == 0) __builtin_amdgcn_s_wait_tensorcnt(0);
#endif
    __syncthreads();
    if (t + 1 < NT) {
#ifdef HAVE_TDM
      if (w == 0)
        tdm_load_tile((unsigned)(size_t)sK[cur ^ 1],
                      Kh + ((size_t)bh * SEQ + (t + 1) * KT) * DIM, KT);
#else
      const uint4* src = (const uint4*)(Kh + ((size_t)bh * SEQ + (t + 1) * KT) * DIM);
      uint4* dst = (uint4*)sK[cur ^ 1];
      for (int i = tid; i < KT * DIM / 8; i += 256) dst[i] = src[i];
#endif
      const int r  = tid >> 2;
      const int dg = (tid & 3) * 16;
      const _Float16* vsrc = Vh + ((size_t)bh * SEQ + (t + 1) * KT + r) * DIM + dg;
#pragma unroll
      for (int i = 0; i < 16; ++i) sVt[cur ^ 1][(dg + i) * KT + r] = vsrc[i];
    }

    unsigned mw[8][2];
#pragma unroll
    for (int v = 0; v < 8; ++v) {
      const int q = q0 + w * 16 + v + 8 * hb;
      uint2 mq = *(const uint2*)&mbits[(size_t)q * SW + t * 2];
      mw[v][0] = mq.x; mw[v][1] = mq.y;
    }

    const _Float16* kt  = sK[cur];
    const _Float16* vtt = sVt[cur];
    _Float16* myP = sP[w];

#pragma unroll
    for (int half = 0; half < 2; ++half) {
#pragma unroll
      for (int sub = 0; sub < 2; ++sub) {
        v8f c = {};
#pragma unroll
        for (int b = 0; b < 2; ++b) {
          v16h bb = *(const v16h*)(kt + (half * 32 + sub * 16 + ln) * DIM + b * 32 + hb * 16);
          c = __builtin_amdgcn_wmma_f32_16x16x32_f16(false, aQ[b], false, bb,
                                                     (short)0, c, false, false);
        }
#pragma unroll
        for (int v = 0; v < 8; ++v) {
          const int bit = (mw[v][half] >> (sub * 16 + ln)) & 1;
          const float p = bit ? __expf(c[v]) * rinv[v] : 0.f;
          const int m = v + 8 * hb;
          const int q = q0 + w * 16 + m;
          // 268 MB streaming output: non-temporal so it doesn't evict K/V/mask
          __builtin_nontemporal_store(
              p, &wout[((size_t)bh * SEQ + q) * SEQ + t * KT + half * 32 + sub * 16 + ln]);
          myP[m * 32 + sub * 16 + ln] = (_Float16)p;  // C-layout -> row-major 16x32
        }
      }

      // P (16x32 fp16) @ V (32x64 fp16), fp32 accumulate: 4 WMMAs per half
      v16h aP;
      {
        const _Float16* prow = myP + ln * 32;
        v8h lo  = *(const v8h*)(prow + hb * 16);
        v8h hi8 = *(const v8h*)(prow + hb * 16 + 8);
#pragma unroll
        for (int j = 0; j < 8; ++j) { aP[j] = lo[j]; aP[8 + j] = hi8[j]; }
      }
#pragma unroll
      for (int db = 0; db < 4; ++db) {
        v16h bv = *(const v16h*)(vtt + (db * 16 + ln) * KT + half * 32 + hb * 16);
        acc[db] = __builtin_amdgcn_wmma_f32_16x16x32_f16(false, aP, false, bv,
                                                         (short)0, acc[db], false, false);
      }
    }
  }

  // ---- store context ----
#pragma unroll
  for (int db = 0; db < 4; ++db)
#pragma unroll
    for (int v = 0; v < 8; ++v) {
      const int q = q0 + w * 16 + v + 8 * hb;
      ctx[((size_t)bh * SEQ + q) * DIM + db * 16 + ln] = acc[db][v];
    }
}

// ---------------------------------------------------------------------------
extern "C" void kernel_launch(void* const* d_in, const int* in_sizes, int n_in,
                              void* d_out, int out_size, void* d_ws, size_t ws_size,
                              hipStream_t stream) {
  const float* Q    = (const float*)d_in[0];
  const float* K    = (const float*)d_in[1];
  const float* V    = (const float*)d_in[2];
  const int*   mask = (const int*)d_in[3];

  float* ctx  = (float*)d_out;                           // [16,2048,64]
  float* wout = (float*)d_out + (size_t)NH * SEQ * DIM;  // [16,2048,2048]

  // workspace: Qh | Kh | Vh (4 MB fp16 each) + packed mask (512 KB)
  const size_t seg = (size_t)NH * SEQ * DIM * sizeof(_Float16);  // 4 MB
  if (ws_size < 3 * seg + (size_t)SEQ * SEQ / 8) return;
  char* ws = (char*)d_ws;
  _Float16* Qh    = (_Float16*)(ws);
  _Float16* Kh    = (_Float16*)(ws + seg);
  _Float16* Vh    = (_Float16*)(ws + 2 * seg);
  unsigned* mbits = (unsigned*)(ws + 3 * seg);

  softmax_q_kernel<<<NH, 256, 0, stream>>>(Q, Qh);
  softmax_k_kernel<<<(2 * SEQ * DIM) / 256, 256, 0, stream>>>(K, Kh);
  convert_v_kernel<<<(NH * SEQ * DIM) / 256, 256, 0, stream>>>(V, Vh);
  pack_mask_kernel<<<(SEQ * SEQ) / 256, 256, 0, stream>>>(mask, mbits);

  dim3 grid(SEQ / QT, NH);
  attn_kernel<<<grid, 256, 0, stream>>>(Qh, Kh, Vh, mbits, ctx, wout);
}
